// GCN2Net_10393820857082
// MI455X (gfx1250) — matrix operand order repaired and checked by reading, not compiled, not run on hardware
//
#include <hip/hip_runtime.h>
#include <cmath>

#define NN     50000
#define EE     800000
#define CC     256
#define GG     64
#define FIN    128
#define ALPHA_ 0.1f

typedef __attribute__((ext_vector_type(16))) __bf16 v16bf;
typedef __attribute__((ext_vector_type(8)))  float  v8f;

__device__ __forceinline__ v8f wmma_bf16(v16bf a, v16bf b, v8f c) {
    return __builtin_amdgcn_wmma_f32_16x16x32_bf16(
        false, a, false, b, (short)0, c, false, false);
}

__device__ __forceinline__ unsigned int pack_bf16(float lo, float hi) {
    unsigned int bl = __float_as_uint(lo);
    bl = (bl + 0x7FFFu + ((bl >> 16) & 1u)) >> 16;
    unsigned int bh = __float_as_uint(hi);
    bh = (bh + 0x7FFFu + ((bh >> 16) & 1u)) >> 16;
    return (bl & 0xFFFFu) | (bh << 16);
}

// order-preserving float->uint key for atomic max
__device__ __forceinline__ unsigned int fkey_enc(float f) {
    unsigned int b = __float_as_uint(f);
    return (b & 0x80000000u) ? ~b : (b | 0x80000000u);
}
__device__ __forceinline__ float fkey_dec(unsigned int k) {
    unsigned int b = (k & 0x80000000u) ? (k ^ 0x80000000u) : ~k;
    return __uint_as_float(b);
}

__device__ __forceinline__ void fadd_dev(float* p, float v) {
    __hip_atomic_fetch_add(p, v, __ATOMIC_RELAXED, __HIP_MEMORY_SCOPE_AGENT);
}
__device__ __forceinline__ void fadd_lds(float* p, float v) {
    __hip_atomic_fetch_add(p, v, __ATOMIC_RELAXED, __HIP_MEMORY_SCOPE_WORKGROUP);
}

// ---------------------------------------------------------------------------
// fill kernel
// ---------------------------------------------------------------------------
__global__ __launch_bounds__(256) void k_fill(float* __restrict__ p, float v, long long n) {
    long long base = (blockIdx.x * 256LL + threadIdx.x) * 4LL;
    if (base + 3 < n) {
        float4 f4 = make_float4(v, v, v, v);
        *(float4*)(p + base) = f4;
    } else if (base < n) {
        for (long long i = base; i < n; ++i) p[i] = v;
    }
}

// ---------------------------------------------------------------------------
// pre-swizzle weights to bf16 WMMA B-operand tile layout.
// dst u32 layout per k-step of 32: [kt][nt (16)][lane (32)][vreg (8)]
// ---------------------------------------------------------------------------
__global__ __launch_bounds__(256) void k_prep(
    const float* __restrict__ src, unsigned int* __restrict__ dst, int K)
{
    int idx = blockIdx.x * 256 + threadIdx.x;
    int units = (K >> 1) * CC;
    if (idx >= units) return;
    int n  = idx & 255;
    int kp = idx >> 8;
    int k  = kp << 1;
    float lo = src[(size_t)k * CC + n];
    float hi = src[(size_t)(k + 1) * CC + n];
    int kt   = k >> 5;
    int kmod = k & 31;
    int lg = (kmod & 8) ? 1 : 0;
    int v  = ((kmod & 7) >> 1) + ((kmod >> 4) << 2);
    int l2 = (n & 15) + 16 * lg;
    int nt = n >> 4;
    dst[kt * 4096 + nt * 256 + l2 * 8 + v] = pack_bf16(lo, hi);
}

// ---------------------------------------------------------------------------
// input GEMM: x0 = h = relu(X[N,128] @ W0[128,256] + b0)
// block: 256 thr (8 waves). M-tile 64, N full 256. K=128 in 4 steps of 32.
// B operand streamed via async global->LDS copy of pre-swizzled bf16 tiles.
// ---------------------------------------------------------------------------
__global__ __launch_bounds__(256) void k_gemm_in(
    const float* __restrict__ x, const unsigned int* __restrict__ Bsw,
    const float* __restrict__ b0, float* __restrict__ x0,
    float* __restrict__ h, int nrows)
{
    __shared__ __align__(32) unsigned int sA[1024];   // 4 tiles * 1KB
    __shared__ __align__(32) unsigned int sB[4096];   // 16 tiles * 1KB

    int tid  = threadIdx.x;
    int lane = tid & 31;
    int w    = tid >> 5;
    int m0   = blockIdx.x * 64;
    unsigned int sb_base = (unsigned int)(unsigned long long)(void*)sB;

    v8f acc[8];
    v8f z8 = {0.f,0.f,0.f,0.f,0.f,0.f,0.f,0.f};
#pragma unroll
    for (int j = 0; j < 8; ++j) acc[j] = z8;

    int mtw = w & 3;
    int ntb = (w >> 2) * 8;

    for (int kt = 0; kt < 4; ++kt) {
        __syncthreads();
        // async-copy pre-swizzled B panel (16KB) into LDS
        const unsigned int* bsrc = Bsw + kt * 4096;
#pragma unroll
        for (int i = 0; i < 4; ++i) {
            int boff = (i * 256 + tid) * 16;
            asm volatile("global_load_async_to_lds_b128 %0, %1, %2"
                         :: "v"(sb_base + boff), "v"(boff), "s"(bsrc)
                         : "memory");
        }
        // stage A (64x32 of X) swizzled to WMMA A layout
#pragma unroll
        for (int i = 0; i < 4; ++i) {
            int u  = i * 256 + tid;
            int kp = u & 15, m = u >> 4;
            int k  = kt * 32 + 2 * kp;
            int gm = m0 + m;
            float lo = 0.f, hi = 0.f;
            if (gm < nrows) {
                const float* p = x + (size_t)gm * FIN + k;
                lo = p[0]; hi = p[1];
            }
            int kmod = 2 * kp;
            int lg   = (kmod & 8) ? 1 : 0;
            int v    = ((kmod & 7) >> 1) + ((kmod >> 4) << 2);
            int l    = (m & 15) + 16 * lg;
            int mt   = m >> 4;
            sA[(mt * 1024 + l * 32 + v * 4) >> 2] = pack_bf16(lo, hi);
        }
        asm volatile("s_wait_asynccnt 0x0" ::: "memory");
        __syncthreads();
        v16bf a = *(const v16bf*)((const char*)sA + mtw * 1024 + lane * 32);
        v16bf bb[8];
#pragma unroll
        for (int j = 0; j < 8; ++j)
            bb[j] = *(const v16bf*)((const char*)sB + (ntb + j) * 1024 + lane * 32);
#pragma unroll
        for (int j = 0; j < 8; ++j)
            acc[j] = wmma_bf16(a, bb[j], acc[j]);
    }

    // epilogue: relu(acc + b0), write x0 and h
    int lhi = lane >> 4;
    int nl  = lane & 15;
#pragma unroll
    for (int j = 0; j < 8; ++j) {
        int n = (ntb + j) * 16 + nl;
        float bn = b0[n];
#pragma unroll
        for (int r = 0; r < 8; ++r) {
            int gm = m0 + mtw * 16 + r + 8 * lhi;
            if (gm < nrows) {
                float val = acc[j][r] + bn;
                val = val > 0.f ? val : 0.f;
                size_t idx = (size_t)gm * CC + n;
                x0[idx] = val;
                h[idx]  = val;
            }
        }
    }
}

// ---------------------------------------------------------------------------
// edge aggregation: agg[dst] += w_e * h[src]  (one thread = 4 channels)
// ---------------------------------------------------------------------------
__global__ __launch_bounds__(256) void k_agg(
    const float* __restrict__ h, const float* __restrict__ ew,
    const int* __restrict__ esrc, const int* __restrict__ edst,
    float* __restrict__ agg)
{
    long long idx = blockIdx.x * 256LL + threadIdx.x;
    int e = (int)(idx >> 6);
    int c = (int)(idx & 63) << 2;
    float wgt = ew[e];
    int s = esrc[e], d = edst[e];
    const float4 hv = *(const float4*)(h + (size_t)s * CC + c);
    float* ap = agg + (size_t)d * CC + c;
    fadd_dev(ap + 0, wgt * hv.x);
    fadd_dev(ap + 1, wgt * hv.y);
    fadd_dev(ap + 2, wgt * hv.z);
    fadd_dev(ap + 3, wgt * hv.w);
}

// ---------------------------------------------------------------------------
// layer GEMM fused: s = 0.9*agg + 0.1*x0 ; out = (1-b)*s + b*(s@Ws)
// t = relu(out + h_prev); accumulate colsum / sumsq for pairnorm
// ---------------------------------------------------------------------------
__global__ __launch_bounds__(256) void k_gemm_layer(
    const float* __restrict__ agg, const float* __restrict__ x0,
    const float* __restrict__ hprev, const unsigned int* __restrict__ Bsw,
    float* __restrict__ t, float* __restrict__ colsum,
    float* __restrict__ sumsq, float betaL, int nrows)
{
    __shared__ __align__(32) unsigned int sA[1024];
    __shared__ __align__(32) unsigned int sB[4096];
    __shared__ float s_col[258];

    int tid  = threadIdx.x;
    int lane = tid & 31;
    int w    = tid >> 5;
    int m0   = blockIdx.x * 64;
    unsigned int sb_base = (unsigned int)(unsigned long long)(void*)sB;

    s_col[tid] = 0.f;
    if (tid == 0) s_col[256] = 0.f;

    v8f acc[8];
    v8f z8 = {0.f,0.f,0.f,0.f,0.f,0.f,0.f,0.f};
#pragma unroll
    for (int j = 0; j < 8; ++j) acc[j] = z8;

    int mtw = w & 3;
    int ntb = (w >> 2) * 8;

    for (int kt = 0; kt < 8; ++kt) {
        __syncthreads();
        // async-copy pre-swizzled B panel (16KB) into LDS
        const unsigned int* bsrc = Bsw + kt * 4096;
#pragma unroll
        for (int i = 0; i < 4; ++i) {
            int boff = (i * 256 + tid) * 16;
            asm volatile("global_load_async_to_lds_b128 %0, %1, %2"
                         :: "v"(sb_base + boff), "v"(boff), "s"(bsrc)
                         : "memory");
        }
        // stage A (s = 0.9*agg + 0.1*x0) swizzled to WMMA A layout
#pragma unroll
        for (int i = 0; i < 4; ++i) {
            int u  = i * 256 + tid;
            int kp = u & 15, m = u >> 4;
            int k  = kt * 32 + 2 * kp;
            int gm = m0 + m;
            float lo = 0.f, hi = 0.f;
            if (gm < nrows) {
                const float* pa = agg + (size_t)gm * CC + k;
                const float* px = x0  + (size_t)gm * CC + k;
                lo = 0.9f * pa[0] + ALPHA_ * px[0];
                hi = 0.9f * pa[1] + ALPHA_ * px[1];
            }
            int kmod = 2 * kp;
            int lg   = (kmod & 8) ? 1 : 0;
            int v    = ((kmod & 7) >> 1) + ((kmod >> 4) << 2);
            int l    = (m & 15) + 16 * lg;
            int mt   = m >> 4;
            sA[(mt * 1024 + l * 32 + v * 4) >> 2] = pack_bf16(lo, hi);
        }
        asm volatile("s_wait_asynccnt 0x0" ::: "memory");
        __syncthreads();
        v16bf a = *(const v16bf*)((const char*)sA + mtw * 1024 + lane * 32);
        v16bf bb[8];
#pragma unroll
        for (int j = 0; j < 8; ++j)
            bb[j] = *(const v16bf*)((const char*)sB + (ntb + j) * 1024 + lane * 32);
#pragma unroll
        for (int j = 0; j < 8; ++j)
            acc[j] = wmma_bf16(a, bb[j], acc[j]);
    }

    // epilogue
    int lhi = lane >> 4;
    int nl  = lane & 15;
    float omb = 1.f - betaL;
    float sst = 0.f;
#pragma unroll
    for (int j = 0; j < 8; ++j) {
        int n = (ntb + j) * 16 + nl;
        float cs = 0.f;
#pragma unroll
        for (int r = 0; r < 8; ++r) {
            int gm = m0 + mtw * 16 + r + 8 * lhi;
            if (gm < nrows) {
                size_t idx = (size_t)gm * CC + n;
                float s  = 0.9f * agg[idx] + ALPHA_ * x0[idx];
                float ov = omb * s + betaL * acc[j][r];
                float tv = ov + hprev[idx];
                tv = tv > 0.f ? tv : 0.f;
                t[idx] = tv;
                cs  += tv;
                sst += tv * tv;
            }
        }
        fadd_lds(&s_col[n], cs);
    }
    // wave-level reduce for sumsq, single LDS atomic per wave
#pragma unroll
    for (int off = 16; off > 0; off >>= 1)
        sst += __shfl_xor(sst, off, 32);
    if (lane == 0) fadd_lds(&s_col[256], sst);

    __syncthreads();
    fadd_dev(&colsum[tid], s_col[tid]);
    if (tid == 0) fadd_dev(sumsq, s_col[256]);
}

// ---------------------------------------------------------------------------
// pairnorm stats: mu[c] = colsum[c]/N ; scale = rsqrt(eps + (sumsq - N|mu|^2)/N)
// ---------------------------------------------------------------------------
__global__ __launch_bounds__(256) void k_stats(
    const float* __restrict__ colsum, const float* __restrict__ sumsq,
    float* __restrict__ mu, float* __restrict__ scale)
{
    __shared__ float red[256];
    int tid = threadIdx.x;
    float m = colsum[tid] * (1.0f / (float)NN);
    mu[tid] = m;
    red[tid] = m * m;
    __syncthreads();
    for (int s = 128; s > 0; s >>= 1) {
        if (tid < s) red[tid] += red[tid + s];
        __syncthreads();
    }
    if (tid == 0) {
        float msq = red[0];
        float var = (sumsq[0] - (float)NN * msq) * (1.0f / (float)NN);
        scale[0] = rsqrtf(1e-5f + var);
    }
}

// h = (t - mu) * scale
__global__ __launch_bounds__(256) void k_norm(
    const float* __restrict__ t, const float* __restrict__ mu,
    const float* __restrict__ scale, float* __restrict__ h, int n4)
{
    int idx = blockIdx.x * 256 + threadIdx.x;
    if (idx >= n4) return;
    size_t base = (size_t)idx * 4;
    int c = (int)(base & (CC - 1));
    float sc = scale[0];
    float4 tv = *(const float4*)(t + base);
    float4 o;
    o.x = (tv.x - mu[c + 0]) * sc;
    o.y = (tv.y - mu[c + 1]) * sc;
    o.z = (tv.z - mu[c + 2]) * sc;
    o.w = (tv.w - mu[c + 3]) * sc;
    *(float4*)(h + base) = o;
}

// ---------------------------------------------------------------------------
// pooling
// ---------------------------------------------------------------------------
__global__ __launch_bounds__(256) void k_pool_init(
    unsigned int* __restrict__ gmaxk, float* __restrict__ gsum,
    float* __restrict__ gcnt)
{
    int idx = blockIdx.x * 256 + threadIdx.x;
    if (idx < GG * CC) { gmaxk[idx] = 0u; gsum[idx] = 0.f; }
    if (idx < GG) gcnt[idx] = 0.f;
}

__global__ __launch_bounds__(256) void k_pool(
    const float* __restrict__ h, const int* __restrict__ batch,
    unsigned int* __restrict__ gmaxk, float* __restrict__ gsum,
    float* __restrict__ gcnt)
{
    long long idx = blockIdx.x * 256LL + threadIdx.x;
    int i = (int)(idx >> 6);
    int c = (int)(idx & 63) << 2;
    int g = batch[i];
    const float4 hv = *(const float4*)(h + (size_t)i * CC + c);
    unsigned int* mk = gmaxk + (size_t)g * CC + c;
    atomicMax(mk + 0, fkey_enc(hv.x));
    atomicMax(mk + 1, fkey_enc(hv.y));
    atomicMax(mk + 2, fkey_enc(hv.z));
    atomicMax(mk + 3, fkey_enc(hv.w));
    float* gs = gsum + (size_t)g * CC + c;
    fadd_dev(gs + 0, hv.x);
    fadd_dev(gs + 1, hv.y);
    fadd_dev(gs + 2, hv.z);
    fadd_dev(gs + 3, hv.w);
    if ((idx & 63) == 0) fadd_dev(&gcnt[g], 1.0f);
}

// ---------------------------------------------------------------------------
// MLP head (single block): z=relu(x2@W1+b1); batchnorm; logits=zn@W2+b2; logsoftmax
// ---------------------------------------------------------------------------
__global__ __launch_bounds__(256) void k_head(
    const unsigned int* __restrict__ gmaxk, const float* __restrict__ gsum,
    const float* __restrict__ gcnt,
    const float* __restrict__ W1, const float* __restrict__ b1,
    const float* __restrict__ W2, const float* __restrict__ b2,
    const float* __restrict__ gamma, const float* __restrict__ betap,
    float* __restrict__ outp)
{
    __shared__ float zsh[GG * 128];
    __shared__ float meanv[128], varv[128];
    __shared__ float lg[GG * 2];
    int tid = threadIdx.x;

    for (int u = tid; u < GG * 128; u += 256) {
        int g = u >> 7, j = u & 127;
        float cnt = gcnt[g];
        cnt = cnt > 1.f ? cnt : 1.f;
        float inv = 1.f / cnt;
        float a = b1[j];
        for (int k = 0; k < CC; ++k)
            a += fkey_dec(gmaxk[(size_t)g * CC + k]) * W1[(size_t)k * 128 + j];
        for (int k = 0; k < CC; ++k)
            a += (gsum[(size_t)g * CC + k] * inv) * W1[(size_t)(CC + k) * 128 + j];
        zsh[u] = a > 0.f ? a : 0.f;
    }
    __syncthreads();
    if (tid < 128) {
        float s = 0.f, s2 = 0.f;
        for (int g = 0; g < GG; ++g) {
            float v = zsh[g * 128 + tid];
            s += v; s2 += v * v;
        }
        float m = s * (1.f / GG);
        meanv[tid] = m;
        varv[tid]  = s2 * (1.f / GG) - m * m;
    }
    __syncthreads();
    if (tid < 128) {
        int g = tid >> 1, o = tid & 1;
        float a = b2[o];
        for (int j = 0; j < 128; ++j) {
            float zn = gamma[j] * (zsh[g * 128 + j] - meanv[j]) *
                       rsqrtf(varv[j] + 1e-5f) + betap[j];
            a += zn * W2[j * 2 + o];
        }
        lg[g * 2 + o] = a;
    }
    __syncthreads();
    if (tid < GG) {
        float a0 = lg[tid * 2], a1 = lg[tid * 2 + 1];
        float mx = fmaxf(a0, a1);
        float lse = mx + logf(expf(a0 - mx) + expf(a1 - mx));
        outp[tid * 2 + 0] = a0 - lse;
        outp[tid * 2 + 1] = a1 - lse;
    }
}

// ---------------------------------------------------------------------------
extern "C" void kernel_launch(void* const* d_in, const int* in_sizes, int n_in,
                              void* d_out, int out_size, void* d_ws, size_t ws_size,
                              hipStream_t stream) {
    (void)in_sizes; (void)n_in; (void)out_size; (void)ws_size;
    const float* x     = (const float*)d_in[0];
    const float* ew    = (const float*)d_in[1];
    const float* W0    = (const float*)d_in[2];
    const float* b0    = (const float*)d_in[3];
    const float* Ws    = (const float*)d_in[4];
    const float* W1    = (const float*)d_in[5];
    const float* b1    = (const float*)d_in[6];
    const float* W2    = (const float*)d_in[7];
    const float* b2    = (const float*)d_in[8];
    const float* gamma = (const float*)d_in[9];
    const float* betap = (const float*)d_in[10];
    const int* esrc    = (const int*)d_in[11];
    const int* edst    = (const int*)d_in[12];
    const int* batch   = (const int*)d_in[13];
    float* outp = (float*)d_out;

    float* ws = (float*)d_ws;
    const size_t NC = (size_t)NN * CC;           // 12.8M floats
    float* x0     = ws;
    float* h      = x0 + NC;
    float* agg    = h + NC;
    float* t      = agg + NC;
    float* colsum = t + NC;                      // 256
    float* sumsq  = colsum + 256;                // 1 (pad 16)
    float* mu     = sumsq + 16;                  // 256
    float* scale  = mu + 256;                    // 1 (pad 16)
    unsigned int* gmaxk = (unsigned int*)(scale + 16);  // G*C
    float* gsum   = (float*)(gmaxk + GG * CC);
    float* gcnt   = gsum + GG * CC;
    unsigned int* W0bf = (unsigned int*)(gcnt + GG);    // 16384 u32
    unsigned int* Wsbf = W0bf + 16384;                  // 8 * 32768 u32

    const int mblocks   = (NN + 63) / 64;                    // 782
    const int aggblocks = (int)(((long long)EE * 64) / 256); // 200000
    const int n4        = (int)(NC / 4);                     // 3,200,000
    const int eltblocks = (n4 + 255) / 256;                  // 12500
    const int fillNC    = (int)((NC / 4 + 255) / 256);

    // pre-swizzle weights to bf16 WMMA layout (once per call)
    k_prep<<<64, 256, 0, stream>>>(W0, W0bf, FIN);
    for (int l = 0; l < 8; ++l)
        k_prep<<<128, 256, 0, stream>>>(Ws + (size_t)l * CC * CC,
                                        Wsbf + (size_t)l * 32768, CC);

    // input layer: x0 = h = relu(X@W0 + b0)
    k_gemm_in<<<mblocks, 256, 0, stream>>>(x, W0bf, b0, x0, h, NN);

    for (int l = 0; l < 8; ++l) {
        float betaL = logf(0.5f / (float)(l + 1) + 1.0f);
        k_fill<<<fillNC, 256, 0, stream>>>(agg, 0.f, (long long)NC);
        k_fill<<<1, 256, 0, stream>>>(colsum, 0.f, 272);
        k_agg<<<aggblocks, 256, 0, stream>>>(h, ew, esrc, edst, agg);
        k_gemm_layer<<<mblocks, 256, 0, stream>>>(agg, x0, h,
                                                  Wsbf + (size_t)l * 32768,
                                                  t, colsum, sumsq, betaL, NN);
        k_stats<<<1, 256, 0, stream>>>(colsum, sumsq, mu, scale);
        k_norm<<<eltblocks, 256, 0, stream>>>(t, mu, scale, h, n4);
    }

    k_pool_init<<<(GG * CC + 255) / 256, 256, 0, stream>>>(gmaxk, gsum, gcnt);
    k_pool<<<(int)(((long long)NN * 64 + 255) / 256), 256, 0, stream>>>(h, batch, gmaxk, gsum, gcnt);
    k_head<<<1, 256, 0, stream>>>(gmaxk, gsum, gcnt, W1, b1, W2, b2, gamma, betap, outp);
}
